// WideDeep_61332132987354
// MI455X (gfx1250) — matrix operand order repaired and checked by reading, not compiled, not run on hardware
//
#include <hip/hip_runtime.h>
#include <hip/hip_bf16.h>

// ---------------- problem constants ----------------
#define BB   16384   // batch
#define FF   26      // onehot fields
#define EE   32      // embedding dim
#define LL   50      // bag length
#define CTC  13      // continuous
#define D0K  909     // mlp in
#define KP0  928     // padded to 32
#define D1N  512
#define D2N  256
#define LEAKY_SLOPE 0.01f

typedef __attribute__((ext_vector_type(16))) _Float16 v16h;
typedef __attribute__((ext_vector_type(8)))  _Float16 v8h;
typedef __attribute__((ext_vector_type(8)))  float    v8f;

// ---------------------------------------------------------------------------
// Weight conversion: f32 [N,K] -> f16 [N,Kp], zero-padded in K.
// ---------------------------------------------------------------------------
__global__ __launch_bounds__(256)
void wd_convert_pad(const float* __restrict__ src, _Float16* __restrict__ dst,
                    int K, int Kp, long total)
{
    long i = (long)blockIdx.x * blockDim.x + threadIdx.x;
    if (i >= total) return;
    int n = (int)(i / Kp);
    int k = (int)(i % Kp);
    float v = (k < K) ? src[(long)n * K + k] : 0.0f;
    dst[i] = (_Float16)v;
}

// ---------------------------------------------------------------------------
// Gather / pool / wide kernel. One 256-thread block (8 waves) per batch row.
//   X[b, 0..831]   = deep[onehot_i[b,f]][e] * onehot_x[b,f]   (col = f*32+e)
//   X[b, 832..863] = sum_l deep[mi1]*mx1
//   X[b, 864..895] = sum_l deep[mi2]*mx2
//   X[b, 896..908] = ctns; X[b, 909..927] = 0
//   wide[b] = 32 * (sum wide_t[idx]*x over all 126 entries)
// ---------------------------------------------------------------------------
__global__ __launch_bounds__(256)
void wd_gather(const int* __restrict__ oi,  const float* __restrict__ ox,
               const int* __restrict__ mi1, const float* __restrict__ mx1,
               const int* __restrict__ mi2, const float* __restrict__ mx2,
               const float* __restrict__ ctns,
               const float* __restrict__ wide_t,
               const float* __restrict__ deep_t,
               _Float16* __restrict__ Xh, float* __restrict__ wide_out)
{
    __shared__ float bag[2][EE];
    __shared__ float wacc;

    const int b    = blockIdx.x;
    const int tid  = threadIdx.x;
    const int wave = tid >> 5;
    const int lane = tid & 31;

    if (tid < 2 * EE) ((float*)bag)[tid] = 0.0f;
    if (tid == 0) wacc = 0.0f;
    __syncthreads();

    // ---- onehot deep embeddings (832 cols), coalesced 128B per field ----
    for (int col = tid; col < FF * EE; col += 256) {
        int f = col >> 5, e = col & 31;
        int   idx = oi[b * FF + f];
        float xv  = ox[b * FF + f];
        float v   = deep_t[(long)idx * EE + e] * xv;
        Xh[(long)b * KP0 + col] = (_Float16)v;
    }

    // ---- tail: ctns + zero pad (cols 896..927) ----
    for (int col = 896 + tid; col < KP0; col += 256) {
        int c = col - 896;
        float v = (c < CTC) ? ctns[b * CTC + c] : 0.0f;
        Xh[(long)b * KP0 + col] = (_Float16)v;
    }

    // ---- multihot bags: waves 0-3 -> bag1, waves 4-7 -> bag2; lane = dim ----
    {
        const int*   mi = (wave < 4) ? mi1 : mi2;
        const float* mx = (wave < 4) ? mx1 : mx2;
        const int  bsel = wave >> 2;
        float acc = 0.0f;
        for (int l = (wave & 3); l < LL; l += 4) {
            int   idx = mi[b * LL + l];
            float xv  = mx[b * LL + l];
            acc += deep_t[(long)idx * EE + lane] * xv;
        }
        atomicAdd(&bag[bsel][lane], acc);   // ds_add_f32
    }

    // ---- wide part: 126 entries over first 126 threads ----
    if (tid < FF + 2 * LL) {
        int idx; float xv;
        if (tid < FF)            { idx = oi[b * FF + tid];            xv = ox[b * FF + tid]; }
        else if (tid < FF + LL)  { idx = mi1[b * LL + (tid - FF)];    xv = mx1[b * LL + (tid - FF)]; }
        else                     { idx = mi2[b * LL + (tid - FF - LL)]; xv = mx2[b * LL + (tid - FF - LL)]; }
        atomicAdd(&wacc, wide_t[idx] * xv);
    }

    __syncthreads();

    if (tid < 2 * EE) {
        int bs = tid >> 5, e = tid & 31;
        Xh[(long)b * KP0 + FF * EE + bs * EE + e] = (_Float16)bag[bs][e];
    }
    if (tid == 0) wide_out[b] = (float)EE * wacc;
}

// ---------------------------------------------------------------------------
// f16 WMMA GEMM:  Y[M,Np](f16) = act( X[M,Kp](f16) @ W[Np,Kp]^T + bias )
// Block = 256 threads = 8 waves (4 M-waves x 2 N-waves).
// Wave computes 64(M) x 32(N) via 4x2 accumulators of v_wmma_f32_16x16x32_f16.
// Block tile = 256(M) x 64(N).  Grid = (M/256, Np/64).
//
// Fragment layouts used here (wave32, contiguous k order; A and B match):
//   A 16x32 f16:  lane m (0..15) holds K k0+0..15 of row m,
//                 lane m+16 holds K k0+16..31              -> contiguous 32B
//   B 32x16 f16:  lane n (0..15) holds K k0+0..15 of col n,
//                 lane n+16 holds K k0+16..31              -> contiguous 32B
//   C/D 16x16 f32: VGPR r, lanes 0-15: (M=r, N=lane); lanes 16-31: (M=r+8).
// ---------------------------------------------------------------------------
__global__ __launch_bounds__(256)
void wd_gemm_f16(const _Float16* __restrict__ X, const _Float16* __restrict__ W,
                 const float* __restrict__ bias, _Float16* __restrict__ Y,
                 int Kp, int Np, int do_leaky)
{
    const int lane  = threadIdx.x & 31;
    const int wave  = threadIdx.x >> 5;
    const int waveM = wave >> 1;              // 0..3
    const int waveN = wave & 1;               // 0..1
    const int m0 = blockIdx.x * 256 + waveM * 64;
    const int n0 = blockIdx.y * 64  + waveN * 32;

    const int lhalf = lane >> 4;              // 0 = lanes 0-15, 1 = lanes 16-31
    const int lmod  = lane & 15;

    v8f acc[4][2] = {};

    for (int k0 = 0; k0 < Kp; k0 += 32) {
        v16h bfrag[2];
#pragma unroll
        for (int ni = 0; ni < 2; ni++) {
            const _Float16* bp =
                W + (size_t)(n0 + ni * 16 + lmod) * Kp + k0 + lhalf * 16;
            bfrag[ni] = *(const v16h*)bp;
        }
#pragma unroll
        for (int mi = 0; mi < 4; mi++) {
            const _Float16* ap =
                X + (size_t)(m0 + mi * 16 + lmod) * Kp + k0 + lhalf * 16;
            v16h afrag = *(const v16h*)ap;
#pragma unroll
            for (int ni = 0; ni < 2; ni++) {
                acc[mi][ni] = __builtin_amdgcn_wmma_f32_16x16x32_f16(
                    false, afrag, false, bfrag[ni],
                    (short)0, acc[mi][ni], false, false);
            }
        }
    }

    // epilogue: bias + optional leaky, store f16
#pragma unroll
    for (int ni = 0; ni < 2; ni++) {
        const int n  = n0 + ni * 16 + lmod;
        const float bn = bias[n];
#pragma unroll
        for (int mi = 0; mi < 4; mi++) {
            const int rbase = m0 + mi * 16 + lhalf * 8;
#pragma unroll
            for (int r = 0; r < 8; r++) {
                float v = acc[mi][ni][r] + bn;
                if (do_leaky) v = (v >= 0.0f) ? v : LEAKY_SLOPE * v;
                Y[(size_t)(rbase + r) * Np + n] = (_Float16)v;
            }
        }
    }
}

// ---------------------------------------------------------------------------
// Final layer: out[b] = sigmoid( dot(Y2[b,:256], w4) + b4 + wide[b] )
// One wave per row, 8 rows per 256-thread block.
// ---------------------------------------------------------------------------
__global__ __launch_bounds__(256)
void wd_final(const _Float16* __restrict__ Y2, const float* __restrict__ w4,
              const float* __restrict__ b4, const float* __restrict__ wide,
              float* __restrict__ out)
{
    const int wave = threadIdx.x >> 5;
    const int lane = threadIdx.x & 31;
    const int b = blockIdx.x * 8 + wave;

    float s = 0.0f;
#pragma unroll
    for (int k = lane; k < D2N; k += 32)
        s += (float)Y2[(size_t)b * D2N + k] * w4[k];

#pragma unroll
    for (int off = 16; off > 0; off >>= 1)
        s += __shfl_xor(s, off, 32);

    if (lane == 0) {
        float z = s + b4[0] + wide[b];
        out[b] = 1.0f / (1.0f + __expf(-z));
    }
}

// ---------------------------------------------------------------------------
extern "C" void kernel_launch(void* const* d_in, const int* in_sizes, int n_in,
                              void* d_out, int out_size, void* d_ws, size_t ws_size,
                              hipStream_t stream)
{
    const int*   oi  = (const int*)  d_in[0];
    const float* ox  = (const float*)d_in[1];
    const int*   mi1 = (const int*)  d_in[2];
    const float* mx1 = (const float*)d_in[3];
    const int*   mi2 = (const int*)  d_in[4];
    const float* mx2 = (const float*)d_in[5];
    const float* ct  = (const float*)d_in[6];
    const float* wt  = (const float*)d_in[7];
    const float* dt  = (const float*)d_in[8];
    const float* w2  = (const float*)d_in[9];
    const float* b2  = (const float*)d_in[10];
    const float* w3  = (const float*)d_in[11];
    const float* b3  = (const float*)d_in[12];
    const float* w4  = (const float*)d_in[13];
    const float* b4  = (const float*)d_in[14];

    char* ws = (char*)d_ws;
    _Float16* Xh  = (_Float16*)ws;  ws += (size_t)BB * KP0 * sizeof(_Float16);
    _Float16* Y1  = (_Float16*)ws;  ws += (size_t)BB * D1N * sizeof(_Float16);
    _Float16* Y2  = (_Float16*)ws;  ws += (size_t)BB * D2N * sizeof(_Float16);
    _Float16* W2h = (_Float16*)ws;  ws += (size_t)D1N * KP0 * sizeof(_Float16);
    _Float16* W3h = (_Float16*)ws;  ws += (size_t)D2N * D1N * sizeof(_Float16);
    float*    wid = (float*)ws;     ws += (size_t)BB * sizeof(float);

    // weight conversion to padded f16
    {
        long t1 = (long)D1N * KP0;
        wd_convert_pad<<<(int)((t1 + 255) / 256), 256, 0, stream>>>(w2, W2h, D0K, KP0, t1);
        long t2 = (long)D2N * D1N;
        wd_convert_pad<<<(int)((t2 + 255) / 256), 256, 0, stream>>>(w3, W3h, D1N, D1N, t2);
    }

    // gather + pool + wide
    wd_gather<<<BB, 256, 0, stream>>>(oi, ox, mi1, mx1, mi2, mx2, ct, wt, dt, Xh, wid);

    // MLP layer 1: [B,928] @ [512,928]^T -> [B,512], leaky
    wd_gemm_f16<<<dim3(BB / 256, D1N / 64), 256, 0, stream>>>(Xh, W2h, b2, Y1, KP0, D1N, 1);

    // MLP layer 2: [B,512] @ [256,512]^T -> [B,256], leaky
    wd_gemm_f16<<<dim3(BB / 256, D2N / 64), 256, 0, stream>>>(Y1, W3h, b3, Y2, D1N, D2N, 1);

    // final dot + sigmoid + wide
    wd_final<<<BB / 8, 256, 0, stream>>>(Y2, w4, b4, wid, (float*)d_out);
}
